// Generator_32014686224395
// MI455X (gfx1250) — matrix operand order repaired and checked by reading, not compiled, not run on hardware
//
#include <hip/hip_runtime.h>

// ---------------------------------------------------------------------------
// MI455X (gfx1250) fused hypernetwork-CA generator.
//   - bf16 WMMA 16x16x32 (v_wmma_f32_16x16x32_bf16) for all per-voxel GEMMs
//   - instance-norm folded into hyper-generated weights (normalized field
//     never materialized)
//   - per-sample 32KB packed weight block staged into LDS with the Tensor
//     Data Mover (tensor_load_to_lds + s_wait_tensorcnt), A-fragments then
//     gathered from LDS; per-wave B-fragments prefetched as one clause.
// ---------------------------------------------------------------------------

typedef __attribute__((ext_vector_type(16))) __bf16 v16bf;
typedef __attribute__((ext_vector_type(8)))  float  v8f;
typedef __attribute__((ext_vector_type(4))) unsigned int v4u;
typedef __attribute__((ext_vector_type(8)))  int  v8i;
typedef __attribute__((ext_vector_type(4)))  int  v4i;

namespace {
constexpr int NB   = 4;       // batch
constexpr int NF   = 32;      // N_FILTER (= FHID = FOUT)
constexpr int V    = 32768;   // 32^3
constexpr int G    = 7;       // 1 + 3 axes * 2 kernel sizes
constexpr int CIN  = 224;     // NF * G
constexpr int KTOT = 16512;   // theta size
constexpr int ZDIM = 512;

// theta element offsets (reference `take` order)
constexpr int OFF_WIN  = 0;
constexpr int OFF_BIN  = 7168;
constexpr int OFF_WMID = 7200;
constexpr int OFF_BMID = 8224;
constexpr int OFF_WOUT = 8256;
constexpr int OFF_BOUT = 9280;
constexpr int OFF_WSH  = 9312;
constexpr int OFF_BSH  = 16480;

// packed per-sample bf16 weight block (exactly 32KB)
constexpr int WP_IN  = 0;      // [32][224]
constexpr int WP_SH  = 7168;   // [32][224]
constexpr int WP_MID = 14336;  // [32][32]
constexpr int WP_OUT = 15360;  // [32][32]
constexpr int WP_SZ  = 16384;  // bf16 elements
}

// --- WMMA fragment helpers -------------------------------------------------
// A (16xK, 16-bit): lane<16 holds K[0..7],K[16..23] of row M=lane;
//                   lane>=16 holds K[8..15],K[24..31]  -> chunks at +0,+32B.
static __device__ inline v16bf frag_a(const __bf16* p) {
  union { uint4 u[2]; v16bf v; } f;
  f.u[0] = *(const uint4*)(p);
  f.u[1] = *(const uint4*)(p + 16);   // +32 bytes
  return f.v;
}
// B (Kx16, 16-bit, column-contiguous): lane<16 holds K[0..7],K[16..23] of col N=lane,
// lane>=16 holds K[8..15],K[24..31]  -> chunks at +0,+32B.
static __device__ inline v16bf frag_b(const __bf16* p) {
  union { uint4 u[2]; v16bf v; } f;
  f.u[0] = *(const uint4*)(p);
  f.u[1] = *(const uint4*)(p + 16);   // +32 bytes
  return f.v;
}
static __device__ inline v8f wmma_bf16(v16bf a, v16bf b, v8f c) {
  return __builtin_amdgcn_wmma_f32_16x16x32_bf16(false, a, false, b,
                                                 (short)0, c, false, false);
}

// --- init: z normalize + tile to 32 channels; theta = hyper_w[y] + hyper_b --
__global__ __launch_bounds__(256) void k_init(const float* __restrict__ z,
                                              const int* __restrict__ y,
                                              const float* __restrict__ hw,
                                              const float* __restrict__ hb,
                                              float* __restrict__ field,
                                              float* __restrict__ theta) {
  int b = blockIdx.x >> 5, c = blockIdx.x & 31, tid = threadIdx.x;
  __shared__ float red[256];
  float s = 0.f;
  for (int i = tid; i < ZDIM; i += 256) { float t = z[b*ZDIM + i]; s += t*t; }
  red[tid] = s; __syncthreads();
  for (int k = 128; k > 0; k >>= 1) { if (tid < k) red[tid] += red[tid+k]; __syncthreads(); }
  float inv = 1.0f / fmaxf(sqrtf(red[0]), 1e-12f);
  float* fc = field + ((size_t)(b*NF + c) << 15);
  for (int v = tid; v < V; v += 256) fc[v] = (v < ZDIM) ? z[b*ZDIM + v]*inv : 0.0f;
  if (c == 0) {
    const float* hrow = hw + (size_t)y[b]*KTOT;
    float* th = theta + (size_t)b*KTOT;
    for (int k = tid; k < KTOT; k += 256) th[k] = hrow[k] + hb[k];
  }
}

// --- SinSobel features (bf16) + per-channel sum/sumsq -----------------------
// block = (b, c, z-slab of 8); LDS halo tile 12x36x36 f32 = 62KB.
// feature channel layout f' = c*7 + g, g in {id, dZ3,dY3,dX3, dZ5,dY5,dX5}.
__global__ __launch_bounds__(256) void k_sobel(const float* __restrict__ field,
                                               __bf16* __restrict__ feats,
                                               float* __restrict__ stats) {
  __shared__ float tile[12*36*36];
  __shared__ float red[14];
  int tid = threadIdx.x;
  int b  = blockIdx.x >> 7;
  int c  = (blockIdx.x >> 2) & 31;
  int zs = blockIdx.x & 3;
  const float* fc = field + ((size_t)(b*NF + c) << 15);
  for (int i = tid; i < 12*36*36; i += 256) {
    int tz = i / 1296, r = i % 1296, ty = r / 36, tx = r % 36;
    int gz = zs*8 + tz - 2, gy = ty - 2, gx = tx - 2;
    float v = 0.f;
    if ((unsigned)gz < 32u && (unsigned)gy < 32u && (unsigned)gx < 32u)
      v = fc[(gz << 10) + (gy << 5) + gx];
    tile[i] = v;
  }
  if (tid < 14) red[tid] = 0.f;
  __syncthreads();

  // sin(linspace(-pi/2,pi/2,k)) / cos(...)+1
  const float d3[3] = {-1.f, 0.f, 1.f};
  const float s3[3] = { 1.f, 2.f, 1.f};
  const float d5[5] = {-1.f, -0.70710678f, 0.f, 0.70710678f, 1.f};
  const float s5[5] = { 1.f,  1.70710678f, 2.f, 1.70710678f, 1.f};

  float ssum[7] = {0,0,0,0,0,0,0}, ssq[7] = {0,0,0,0,0,0,0};
  for (int j = 0; j < 32; ++j) {
    int idx = tid + (j << 8);                 // 0..8191 within slab
    int lz = idx >> 10, ly = (idx >> 5) & 31, lx = idx & 31;
    const float* ctr = &tile[(lz+2)*1296 + (ly+2)*36 + (lx+2)];
    float a[7] = {ctr[0], 0,0,0,0,0,0};
    #pragma unroll
    for (int dz = -1; dz <= 1; ++dz)
      #pragma unroll
      for (int dy = -1; dy <= 1; ++dy) {
        float w_ds = d3[dz+1]*s3[dy+1];
        float w_sd = s3[dz+1]*d3[dy+1];
        float w_ss = s3[dz+1]*s3[dy+1];
        #pragma unroll
        for (int dx = -1; dx <= 1; ++dx) {
          float v = ctr[dz*1296 + dy*36 + dx];
          a[1] += v * (w_ds * s3[dx+1]);   // deriv along D(z)
          a[2] += v * (w_sd * s3[dx+1]);   // deriv along H(y)
          a[3] += v * (w_ss * d3[dx+1]);   // deriv along W(x)
        }
      }
    #pragma unroll
    for (int dz = -2; dz <= 2; ++dz)
      #pragma unroll
      for (int dy = -2; dy <= 2; ++dy) {
        float w_ds = d5[dz+2]*s5[dy+2];
        float w_sd = s5[dz+2]*d5[dy+2];
        float w_ss = s5[dz+2]*s5[dy+2];
        #pragma unroll
        for (int dx = -2; dx <= 2; ++dx) {
          float v = ctr[dz*1296 + dy*36 + dx];
          a[4] += v * (w_ds * s5[dx+2]);
          a[5] += v * (w_sd * s5[dx+2]);
          a[6] += v * (w_ss * d5[dx+2]);
        }
      }
    int vox = ((zs*8 + lz) << 10) + (ly << 5) + lx;
    __bf16* fo = feats + ((size_t)b*V + vox)*CIN + c*G;
    #pragma unroll
    for (int g = 0; g < 7; ++g) {
      fo[g] = (__bf16)a[g];
      ssum[g] += a[g];
      ssq[g]  += a[g]*a[g];
    }
  }
  #pragma unroll
  for (int g = 0; g < 7; ++g) { atomicAdd(&red[g], ssum[g]); atomicAdd(&red[7+g], ssq[g]); }
  __syncthreads();
  if (tid < 7)       atomicAdd(&stats[b*2*CIN + c*G + tid], red[tid]);
  else if (tid < 14) atomicAdd(&stats[b*2*CIN + CIN + c*G + (tid-7)], red[tid]);
}

// --- fold instance-norm into hyper weights, quantize to packed bf16 ---------
// feature col f' = c*7+g  maps to reference channel forig = g*32 + c.
__global__ __launch_bounds__(256) void k_fold(const float* __restrict__ theta,
    const float* __restrict__ stats,
    __bf16* __restrict__ wPack,
    float* __restrict__ bIn, float* __restrict__ bMid, float* __restrict__ bOS) {
  int b = blockIdx.x, tid = threadIdx.x;
  const float* th = theta + (size_t)b*KTOT;
  const float* st = stats + b*2*CIN;
  __shared__ float sm[CIN], sc[CIN];
  if (tid < CIN) {
    float m   = st[tid] * (1.0f/V);
    float var = fmaxf(st[CIN + tid] * (1.0f/V) - m*m, 0.0f);
    sm[tid] = m;
    sc[tid] = rsqrtf(var + 1e-5f);
  }
  __syncthreads();
  for (int e = tid; e < WP_SZ; e += 256) {
    float val;
    if (e < WP_SH) {
      int o = e / CIN, f = e % CIN;
      int forig = (f % G)*NF + f / G;
      val = th[OFF_WIN + o*CIN + forig] * sc[f];
    } else if (e < WP_MID) {
      int e2 = e - WP_SH;
      int o = e2 / CIN, f = e2 % CIN;
      int forig = (f % G)*NF + f / G;
      val = th[OFF_WSH + o*CIN + forig] * sc[f];
    } else if (e < WP_OUT) {
      val = th[OFF_WMID + (e - WP_MID)];
    } else {
      val = th[OFF_WOUT + (e - WP_OUT)];
    }
    wPack[(size_t)b*WP_SZ + e] = (__bf16)val;
  }
  if (tid < NF) {
    int o = tid;
    float c1 = 0.f, c2 = 0.f;
    for (int f = 0; f < CIN; ++f) {
      int forig = (f % G)*NF + f / G;
      float smf = sc[f]*sm[f];
      c1 += th[OFF_WIN + o*CIN + forig] * smf;
      c2 += th[OFF_WSH + o*CIN + forig] * smf;
    }
    bIn[b*NF + o]  = th[OFF_BIN  + o] - c1;
    bMid[b*NF + o] = th[OFF_BMID + o];
    bOS[b*NF + o]  = th[OFF_BOUT + o] + th[OFF_BSH + o] - c2;   // out + short
  }
}

// --- fused per-voxel MLP via WMMA; residual update of field -----------------
// grid = NB * (V/128); 8 waves/block; wave 0 TDM-stages the 32KB packed
// weight block into LDS; each wave owns 16 voxels: 2x7+2x7+2+2 = 32 WMMAs.
__global__ __launch_bounds__(256) void k_mlp(const __bf16* __restrict__ feats,
    const __bf16* __restrict__ wPack,
    const float* __restrict__ bIn, const float* __restrict__ bMid,
    const float* __restrict__ bOS, float* __restrict__ field) {
  __shared__ __bf16 sW[WP_SZ];       // 32KB packed weights for this sample
  __shared__ __bf16 lh[8*16*32];     // per-wave 16 voxels x 32 hidden ch
  int tid  = threadIdx.x;
  int lane = tid & 31, wv = tid >> 5;
  int n    = lane & 15, half = lane >> 4;
  int b    = blockIdx.x >> 8;
  int tilei = blockIdx.x & 255;
  int vox  = tilei*128 + wv*16 + n;                 // this lane's voxel column

  // ---- TDM: DMA the 32KB weight block global -> LDS (1-D tile, 8B elems) --
  if (wv == 0) {
    unsigned long long ga =
        (unsigned long long)(size_t)(wPack + (size_t)b*WP_SZ);
    unsigned ldsOff =
        (unsigned)(size_t)(__attribute__((address_space(3))) __bf16*)sW;
    // D# group0: count=1 | lds_addr | global_addr[56:0] | type=2
    v4u g0 = { 1u,
               ldsOff,
               (unsigned)(ga & 0xffffffffu),
               (unsigned)((ga >> 32) & 0x01ffffffu) | (2u << 30) };
    // D# group1: data_size=3 (8B); tensor_dim0 = tile_dim0 = 4096 elems;
    //            tensor_dim1 = 1; tensor_dim0_stride = 4096.
    v8i g1 = { (int)(3u << 16),
               (int)(4096u << 16),     // tensor_dim0[15:0] @ bits 63:48
               (int)(1u << 16),        // tensor_dim1[15:0] @ bits 95:80
               (int)(4096u << 16),     // tile_dim0 @ bits 127:112
               0,                      // tile_dim1/2 unused
               4096,                   // tensor_dim0_stride[31:0]
               0, 0 };
    v4i z4 = {0, 0, 0, 0};
    v8i z8 = {0, 0, 0, 0, 0, 0, 0, 0};
    __builtin_amdgcn_tensor_load_to_lds(g0, g1, z4, z4, z8, 0);
    __builtin_amdgcn_s_wait_tensorcnt(0);
  }
  __syncthreads();

  // ---- prefetch all 7 B-fragments (this lane's voxel column) as one clause
  const __bf16* xcol = feats + ((size_t)b*V + vox)*CIN + half*8;
  v16bf bfr[7];
  #pragma unroll
  for (int kk = 0; kk < 7; ++kk) bfr[kk] = frag_b(xcol + kk*32);

  const __bf16* AIn = sW + WP_IN;
  const __bf16* ASh = sW + WP_SH;
  const __bf16* AMd = sW + WP_MID;
  const __bf16* AOt = sW + WP_OUT;
  int ar0 = n, ar1 = 16 + n;                        // A rows for M-tiles 0/1

  v8f aH0 = {0,0,0,0,0,0,0,0}, aH1 = {0,0,0,0,0,0,0,0};
  v8f aS0 = {0,0,0,0,0,0,0,0}, aS1 = {0,0,0,0,0,0,0,0};
  #pragma unroll
  for (int kk = 0; kk < 7; ++kk) {                  // K = 224 = 7 * 32
    aH0 = wmma_bf16(frag_a(AIn + ar0*CIN + kk*32 + half*8), bfr[kk], aH0);
    aH1 = wmma_bf16(frag_a(AIn + ar1*CIN + kk*32 + half*8), bfr[kk], aH1);
    aS0 = wmma_bf16(frag_a(ASh + ar0*CIN + kk*32 + half*8), bfr[kk], aS0);
    aS1 = wmma_bf16(frag_a(ASh + ar1*CIN + kk*32 + half*8), bfr[kk], aS1);
  }

  // H1 = lrelu(W_in' x + b_in') -> LDS (in-wave LDS ordering: no barrier)
  const float* bInb = bIn + b*NF;
  __bf16* hcol = &lh[wv*512 + n*32];
  #pragma unroll
  for (int r = 0; r < 8; ++r) {
    int o0 = r + half*8, o1 = 16 + r + half*8;
    float h0 = aH0[r] + bInb[o0]; h0 = fmaxf(h0, 0.2f*h0);
    float h1 = aH1[r] + bInb[o1]; h1 = fmaxf(h1, 0.2f*h1);
    hcol[o0] = (__bf16)h0; hcol[o1] = (__bf16)h1;
  }

  // H2 = lrelu(W_mid H1 + b_mid)
  v8f a20 = {0,0,0,0,0,0,0,0}, a21 = {0,0,0,0,0,0,0,0};
  {
    v16bf bh = frag_b(&lh[wv*512 + n*32 + half*8]);
    a20 = wmma_bf16(frag_a(AMd + ar0*NF + half*8), bh, a20);
    a21 = wmma_bf16(frag_a(AMd + ar1*NF + half*8), bh, a21);
  }
  const float* bMb = bMid + b*NF;
  #pragma unroll
  for (int r = 0; r < 8; ++r) {
    int o0 = r + half*8, o1 = 16 + r + half*8;
    float h0 = a20[r] + bMb[o0]; h0 = fmaxf(h0, 0.2f*h0);
    float h1 = a21[r] + bMb[o1]; h1 = fmaxf(h1, 0.2f*h1);
    hcol[o0] = (__bf16)h0; hcol[o1] = (__bf16)h1;   // overwrite H1 (in-order)
  }

  // O = W_out H2 + (W_sh' x) + (b_out + b_sh'); field += 0.1 * O
  v8f a30 = {0,0,0,0,0,0,0,0}, a31 = {0,0,0,0,0,0,0,0};
  {
    v16bf bh = frag_b(&lh[wv*512 + n*32 + half*8]);
    a30 = wmma_bf16(frag_a(AOt + ar0*NF + half*8), bh, a30);
    a31 = wmma_bf16(frag_a(AOt + ar1*NF + half*8), bh, a31);
  }
  const float* bOb = bOS + b*NF;
  float* fb = field + ((size_t)(b*NF) << 15) + vox;
  #pragma unroll
  for (int r = 0; r < 8; ++r) {
    int o0 = r + half*8, o1 = 16 + r + half*8;
    float v0 = a30[r] + aS0[r] + bOb[o0];
    float v1 = a31[r] + aS1[r] + bOb[o1];
    fb[(size_t)o0 << 15] += 0.1f * v0;
    fb[(size_t)o1 << 15] += 0.1f * v1;
  }
}

// --- final latent: channel mean --------------------------------------------
__global__ __launch_bounds__(256) void k_lat(const float* __restrict__ field,
                                             float* __restrict__ out) {
  int idx = blockIdx.x*256 + threadIdx.x;       // NB*V threads
  int b = idx >> 15, v = idx & (V - 1);
  const float* fb = field + ((size_t)(b*NF) << 15) + v;
  float s = 0.f;
  #pragma unroll
  for (int c = 0; c < NF; ++c) s += fb[(size_t)c << 15];
  out[idx] = s * (1.0f/NF);
}

// ---------------------------------------------------------------------------
extern "C" void kernel_launch(void* const* d_in, const int* in_sizes, int n_in,
                              void* d_out, int out_size, void* d_ws, size_t ws_size,
                              hipStream_t stream) {
  (void)in_sizes; (void)n_in; (void)out_size; (void)ws_size;
  const float* z  = (const float*)d_in[0];
  const int*   y  = (const int*)d_in[1];
  const float* hw = (const float*)d_in[2];
  const float* hb = (const float*)d_in[3];

  char* p = (char*)d_ws;
  auto carve = [&](size_t bytes) -> void* {
    char* q = p; p += (bytes + 255) & ~(size_t)255; return (void*)q;
  };
  float*  field = (float*) carve((size_t)NB*NF*V*4);      // 16.8 MB
  __bf16* feats = (__bf16*)carve((size_t)NB*V*CIN*2);     // 58.7 MB
  float*  stats = (float*) carve((size_t)NB*2*CIN*4);
  float*  theta = (float*) carve((size_t)NB*KTOT*4);
  __bf16* wPack = (__bf16*)carve((size_t)NB*WP_SZ*2);     // 4 x 32KB
  float*  bIn   = (float*) carve((size_t)NB*NF*4);
  float*  bMid  = (float*) carve((size_t)NB*NF*4);
  float*  bOS   = (float*) carve((size_t)NB*NF*4);

  k_init<<<NB*NF, 256, 0, stream>>>(z, y, hw, hb, field, theta);
  for (int it = 0; it < 4; ++it) {
    (void)hipMemsetAsync(stats, 0, (size_t)NB*2*CIN*4, stream);
    k_sobel<<<NB*NF*4, 256, 0, stream>>>(field, feats, stats);
    k_fold<<<NB, 256, 0, stream>>>(theta, stats, wPack, bIn, bMid, bOS);
    k_mlp<<<NB*(V/128), 256, 0, stream>>>(feats, wPack, bIn, bMid, bOS, field);
  }
  k_lat<<<(NB*V)/256, 256, 0, stream>>>(field, (float*)d_out);
}
